// SimAttnPE1_24739011625739
// MI455X (gfx1250) — matrix operand order repaired and checked by reading, not compiled, not run on hardware
//
#include <hip/hip_runtime.h>
#include <hip/hip_bf16.h>

// ---------------------------------------------------------------------------
// SimAttnPE1 pipeline for MI455X (gfx1250, wave32, WMMA bf16 16x16x32).
//
// Roofline: dominant traffic is fc_w (314.6 MB fp32) -> 13.5us at 23.3 TB/s.
// bf16 WMMA keeps every GEMM memory-bound; fc_w is converted fp32->bf16 in
// registers inside the fc GEMM so it is read from HBM exactly once.
// Deterministic: split-K partials in workspace + fixed-order reduction
// (no floating-point atomics anywhere).
// ---------------------------------------------------------------------------

#define VSZ 50000
#define LSEQ 512
#define DDIM 300
#define CDIM 512
#define BATCH 64
#define DP 320                 // D padded to multiple of 32 for WMMA K
#define KFC 153600L            // C*D, multiple of 32 -> no pad needed flat
#define NSPLIT 96              // fc split-K factor (4800 k-steps / 96 = 50)
#define KSTEPS_PER_SPLIT 50

typedef __attribute__((ext_vector_type(16))) __bf16 v16bf;
typedef __attribute__((ext_vector_type(8)))  __bf16 v8bf;
typedef __attribute__((ext_vector_type(8)))  float  v8f;
typedef __attribute__((ext_vector_type(16))) float  v16f;

union AFrag { v16bf v; v8bf h8[2]; };

// ---------------------------------------------------------------------------
// Kernel 1: conv_w [C,D] f32 -> cw [C,DP] bf16 (zero pad tail)
// ---------------------------------------------------------------------------
__global__ void conv_to_bf16(const float* __restrict__ cwf, __bf16* __restrict__ cw) {
    int i = blockIdx.x * 256 + threadIdx.x;          // 0 .. C*DP-1
    int c = i / DP, d = i - c * DP;
    cw[i] = (__bf16)(d < DDIM ? cwf[c * DDIM + d] : 0.0f);
}

// ---------------------------------------------------------------------------
// Kernel 2: embedding gather + positional-encoding blend -> h [B,L,DP] bf16
// ---------------------------------------------------------------------------
__global__ __launch_bounds__(DP) void embed_pe(const int* __restrict__ x,
                                               const float* __restrict__ ew,
                                               const float* __restrict__ coefp,
                                               const float* __restrict__ pe,
                                               __bf16* __restrict__ h) {
    int bl = blockIdx.x;                 // b*L + l
    int l  = bl & (LSEQ - 1);
    int t  = threadIdx.x;                // 0..319
    float coef = coefp[0];
    int idx = x[bl];
    float v = 0.0f;
    if (t < DDIM)
        v = (1.0f - coef) * ew[(size_t)idx * DDIM + t] + coef * pe[(size_t)l * DDIM + t];
    h[(size_t)bl * DP + t] = (__bf16)v;
}

// ---------------------------------------------------------------------------
// Kernel 3: LDS-tiled transpose h [B,L,DP] -> ht [B,DP,L]  (for ctx B-operand)
// ---------------------------------------------------------------------------
__global__ void transpose_h(const __bf16* __restrict__ h, __bf16* __restrict__ ht) {
    __shared__ __bf16 tile[32][33];
    int b = blockIdx.z, l0 = blockIdx.y * 32, d0 = blockIdx.x * 32;
    int tx = threadIdx.x, ty = threadIdx.y;          // (32, 8)
    for (int i = ty; i < 32; i += 8)
        tile[i][tx] = h[((size_t)(b * LSEQ + l0 + i)) * DP + d0 + tx];
    __syncthreads();
    for (int i = ty; i < 32; i += 8)
        ht[((size_t)(b * DP + d0 + i)) * LSEQ + l0 + tx] = tile[tx][i];
}

// ---------------------------------------------------------------------------
// Kernel 4: sim = h @ conv_w^T fused with softmax over L -> p [B,C,L] bf16
// One block per (b, 16-row c slab). 8 waves x 4 l-tiles, K=320 (10 WMMA).
// conv_b omitted: constant per (b,c) row => cancels in softmax (and is zero).
// ---------------------------------------------------------------------------
__global__ __launch_bounds__(256) void sim_softmax(const __bf16* __restrict__ h,
                                                   const __bf16* __restrict__ cw,
                                                   __bf16* __restrict__ p) {
    __shared__ float sc[16][LSEQ];                   // 32 KB
    int blk  = blockIdx.x;
    int b    = blk >> 5;                             // / (C/16)
    int c0   = (blk & 31) << 4;
    int lane = threadIdx.x & 31, wave = threadIdx.x >> 5;
    int half = lane >> 4, nn = lane & 15;

    const __bf16* arow = cw + (size_t)(c0 + nn) * DP;        // A: m = lane&15
    for (int t = 0; t < 4; ++t) {
        int l0 = (wave * 4 + t) * 16;
        const __bf16* brow = h + ((size_t)(b * LSEQ + l0 + nn)) * DP + 16 * half;
        v8f acc = {};
        for (int ks = 0; ks < 10; ++ks) {
            int k0 = ks * 32;
            AFrag a;
            a.v = *(const v16bf*)(arow + k0 + 16 * half);
            v16bf bv = *(const v16bf*)(brow + k0);
            acc = __builtin_amdgcn_wmma_f32_16x16x32_bf16(
                false, a.v, false, bv, (short)0, acc, false, false);
        }
        for (int i = 0; i < 8; ++i)
            sc[i + 8 * half][l0 + nn] = acc[i];              // D: m=i+8*half, n=lane&15
    }
    __syncthreads();

    // softmax over L for each of the 16 c-rows (one wave per row, 2 rows/wave)
    for (int r = wave; r < 16; r += 8) {
        float vals[16];
        float mx = -3.0e38f;
        for (int j = 0; j < 16; ++j) {
            vals[j] = sc[r][lane + 32 * j];
            mx = fmaxf(mx, vals[j]);
        }
        for (int off = 16; off; off >>= 1) mx = fmaxf(mx, __shfl_xor(mx, off, 32));
        float sum = 0.0f;
        for (int j = 0; j < 16; ++j) { vals[j] = __expf(vals[j] - mx); sum += vals[j]; }
        for (int off = 16; off; off >>= 1) sum += __shfl_xor(sum, off, 32);
        float inv = 1.0f / sum;
        __bf16* prow = p + ((size_t)(b * CDIM + c0 + r)) * LSEQ;
        for (int j = 0; j < 16; ++j)
            prow[lane + 32 * j] = (__bf16)(vals[j] * inv);
    }
}

// ---------------------------------------------------------------------------
// Kernel 5: ctx = p @ h -> ctx [B, C*D] bf16 flat (k = c*D + d), K = L = 512
// One block per (b, 16-row c slab); waves cover 19 d-tiles (d padded reads
// land in the zeroed DP tail; stores guarded to d < 300).
// ---------------------------------------------------------------------------
__global__ __launch_bounds__(256) void ctx_gemm(const __bf16* __restrict__ p,
                                                const __bf16* __restrict__ ht,
                                                __bf16* __restrict__ ctx) {
    int blk  = blockIdx.x;
    int b    = blk >> 5;
    int c0   = (blk & 31) << 4;
    int lane = threadIdx.x & 31, wave = threadIdx.x >> 5;
    int half = lane >> 4, nn = lane & 15;

    const __bf16* arow = p + ((size_t)(b * CDIM + c0 + nn)) * LSEQ;
    for (int dt = wave; dt < 19; dt += 8) {
        int d0 = dt * 16;
        const __bf16* brow = ht + ((size_t)(b * DP + d0 + nn)) * LSEQ + 16 * half;
        v8f acc = {};
        for (int ks = 0; ks < 16; ++ks) {
            int k0 = ks * 32;
            AFrag a;
            a.v = *(const v16bf*)(arow + k0 + 16 * half);
            v16bf bv = *(const v16bf*)(brow + k0);
            acc = __builtin_amdgcn_wmma_f32_16x16x32_bf16(
                false, a.v, false, bv, (short)0, acc, false, false);
        }
        int d = d0 + nn;
        if (d < DDIM) {
            for (int i = 0; i < 8; ++i) {
                int m = i + 8 * half;
                ctx[(size_t)b * KFC + (size_t)(c0 + m) * DDIM + d] = (__bf16)acc[i];
            }
        }
    }
}

// ---------------------------------------------------------------------------
// Kernel 6: fc split-K partials. One wave per block. 4x4 output tiles
// (all 4 M-tiles x a 4-N-tile group) so fc_w streams from HBM exactly once.
// fc_w fp32 is converted to bf16 in registers (no HBM round-trip).
// ---------------------------------------------------------------------------
__global__ __launch_bounds__(32) void fc_partial(const __bf16* __restrict__ ctx,
                                                 const float* __restrict__ fcw,
                                                 float* __restrict__ part) {
    int g    = blockIdx.y;                   // 0..7 : group of 4 n-tiles
    int s    = blockIdx.x;                   // 0..NSPLIT-1 : K split
    int lane = threadIdx.x;
    int half = lane >> 4, nn = lane & 15;

    v8f acc[4][4] = {};
    long kbase = (long)s * KSTEPS_PER_SPLIT * 32;
    for (int ks = 0; ks < KSTEPS_PER_SPLIT; ++ks) {
        long k0 = kbase + (long)ks * 32;
        v16bf a[4];
        for (int mt = 0; mt < 4; ++mt) {
            const __bf16* row = ctx + (size_t)(mt * 16 + nn) * KFC;
            AFrag ua;
            ua.v = *(const v16bf*)(row + k0 + 16 * half);
            a[mt] = ua.v;
        }
        for (int nt = 0; nt < 4; ++nt) {
            int c = g * 64 + nt * 16 + nn;
            const float* wrow = fcw + (size_t)c * KFC + k0 + 16 * half;
            v16f wf = *(const v16f*)wrow;            // 64B fp32 load
            v16bf bw;
            for (int i = 0; i < 16; ++i) bw[i] = (__bf16)wf[i];   // cvt in regs
            for (int mt = 0; mt < 4; ++mt)
                acc[mt][nt] = __builtin_amdgcn_wmma_f32_16x16x32_bf16(
                    false, a[mt], false, bw, (short)0, acc[mt][nt], false, false);
        }
    }
    // partials: part[s][b][c] (disjoint per block -> no atomics, deterministic)
    float* ps = part + (size_t)s * (BATCH * CDIM);
    for (int mt = 0; mt < 4; ++mt)
        for (int nt = 0; nt < 4; ++nt) {
            int c = g * 64 + nt * 16 + nn;
            for (int i = 0; i < 8; ++i) {
                int b = mt * 16 + i + 8 * half;
                ps[b * CDIM + c] = acc[mt][nt][i];
            }
        }
}

// ---------------------------------------------------------------------------
// Kernel 7: fixed-order split-K reduction + bias -> out [B, C] f32
// ---------------------------------------------------------------------------
__global__ void fc_reduce(const float* __restrict__ part,
                          const float* __restrict__ fc_b,
                          float* __restrict__ out) {
    int idx = blockIdx.x * 256 + threadIdx.x;        // 0 .. B*C-1
    int c = idx & (CDIM - 1);
    float s = fc_b[c];
    for (int i = 0; i < NSPLIT; ++i)
        s += part[(size_t)i * (BATCH * CDIM) + idx];
    out[idx] = s;
}

// ---------------------------------------------------------------------------
extern "C" void kernel_launch(void* const* d_in, const int* in_sizes, int n_in,
                              void* d_out, int out_size, void* d_ws, size_t ws_size,
                              hipStream_t stream) {
    (void)in_sizes; (void)n_in; (void)out_size; (void)ws_size;
    const int*   x       = (const int*)d_in[0];
    const float* embed_w = (const float*)d_in[1];
    const float* coef    = (const float*)d_in[2];
    const float* pe      = (const float*)d_in[3];
    const float* conv_w  = (const float*)d_in[4];
    // d_in[5] = conv_b : constant per softmax row => cancels (and is zero)
    const float* fc_w    = (const float*)d_in[6];
    const float* fc_b    = (const float*)d_in[7];
    float*       out     = (float*)d_out;

    // workspace partition (256-B aligned), total ~108 MB
    char* ws = (char*)d_ws;
    size_t off = 0;
    auto take = [&](size_t bytes) -> void* {
        void* q = ws + off;
        off += (bytes + 255) & ~(size_t)255;
        return q;
    };
    __bf16* h    = (__bf16*)take((size_t)BATCH * LSEQ * DP * 2);   // 21.0 MB
    __bf16* ht   = (__bf16*)take((size_t)BATCH * DP * LSEQ * 2);   // 21.0 MB
    __bf16* cw   = (__bf16*)take((size_t)CDIM * DP * 2);           // 0.33 MB
    __bf16* pbuf = (__bf16*)take((size_t)BATCH * CDIM * LSEQ * 2); // 33.6 MB
    __bf16* ctx  = (__bf16*)take((size_t)BATCH * KFC * 2);         // 19.7 MB
    float*  part = (float*)take((size_t)NSPLIT * BATCH * CDIM * 4);// 12.6 MB

    conv_to_bf16<<<(CDIM * DP) / 256, 256, 0, stream>>>(conv_w, cw);
    embed_pe<<<BATCH * LSEQ, DP, 0, stream>>>(x, embed_w, coef, pe, h);
    transpose_h<<<dim3(DP / 32, LSEQ / 32, BATCH), dim3(32, 8), 0, stream>>>(h, ht);
    sim_softmax<<<BATCH * (CDIM / 16), 256, 0, stream>>>(h, cw, pbuf);
    ctx_gemm<<<BATCH * (CDIM / 16), 256, 0, stream>>>(pbuf, ht, ctx);
    fc_partial<<<dim3(NSPLIT, 8), 32, 0, stream>>>(ctx, fc_w, part);
    fc_reduce<<<(BATCH * CDIM) / 256, 256, 0, stream>>>(part, fc_b, out);
}
